// MultiHeadAttention_62294205661812
// MI455X (gfx1250) — matrix operand order repaired and checked by reading, not compiled, not run on hardware
//
#include <hip/hip_runtime.h>
#include <hip/hip_bf16.h>

typedef __attribute__((ext_vector_type(16))) _Float16 v16h;
typedef __attribute__((ext_vector_type(8)))  float    v8f;

#define B_  8
#define S_  2048
#define D_  128
#define H_  4
#define DH_ 32
#define KT_ 64                             /* keys per flash iteration */
#define NEG_MASK (-4294967295.0f)          /* -2^32 + 1 */
#define ISQ 0.1767766952966369f            /* 1/sqrt(32) */

union AFrag { v16h h; unsigned u[8]; };

static __device__ __forceinline__ v8f wmma_f16(v16h a, v16h b, v8f c) {
  // D = A(16x32 f16) * B(32x16 f16) + C(16x16 f32)
  return __builtin_amdgcn_wmma_f32_16x16x32_f16(false, a, false, b, (short)0, c,
                                                false, false);
}

// ---------------------------------------------------------------------------
// Kernel 1: QKV projection.  out = X @ W^T + b, written as f16.
//   mode 0 -> Qh [B,H,S,32], mode 1 -> Kh [B,H,S,32], mode 2 -> Vt [B,H,32,S]
// ---------------------------------------------------------------------------
__global__ __launch_bounds__(256) void qkv_proj_kernel(
    const float* __restrict__ query, const float* __restrict__ key,
    const float* __restrict__ value,
    const float* __restrict__ WQ_w, const float* __restrict__ WQ_b,
    const float* __restrict__ WK_w, const float* __restrict__ WK_b,
    const float* __restrict__ WV_w, const float* __restrict__ WV_b,
    _Float16* __restrict__ Qh, _Float16* __restrict__ Kh,
    _Float16* __restrict__ Vt)
{
  __shared__ alignas(32) _Float16 Xs[128 * 128];
  __shared__ alignas(32) _Float16 Ws[128 * 128];

  const int mode = blockIdx.y;
  const float* X; const float* W; const float* bias;
  if (mode == 0)      { X = query; W = WQ_w; bias = WQ_b; }
  else if (mode == 1) { X = key;   W = WK_w; bias = WK_b; }
  else                { X = value; W = WV_w; bias = WV_b; }

  const int g  = blockIdx.x * 128;   // global starting row of the tile
  const int b  = g / S_;
  const int s0 = g % S_;

  // Stage X tile (128x128) and full W (128x128) into LDS as f16.
  const float4* Xv = (const float4*)(X + (size_t)g * D_);
  const float4* Wv = (const float4*)W;
  for (int i = threadIdx.x; i < (128 * 128) / 4; i += 256) {
    float4 xv = Xv[i];
    Xs[i * 4 + 0] = (_Float16)xv.x; Xs[i * 4 + 1] = (_Float16)xv.y;
    Xs[i * 4 + 2] = (_Float16)xv.z; Xs[i * 4 + 3] = (_Float16)xv.w;
    float4 wv = Wv[i];
    Ws[i * 4 + 0] = (_Float16)wv.x; Ws[i * 4 + 1] = (_Float16)wv.y;
    Ws[i * 4 + 2] = (_Float16)wv.z; Ws[i * 4 + 3] = (_Float16)wv.w;
  }
  __syncthreads();

  const int wave = threadIdx.x >> 5;
  const int lane = threadIdx.x & 31;
  const int l15  = lane & 15;
  const int hi   = lane >> 4;
  const int row0 = wave * 16;        // this wave's 16 output rows

  // Hoist the wave's A-fragments (invariant across output-column tiles).
  AFrag a[4];
#pragma unroll
  for (int kk = 0; kk < 4; ++kk) {
#pragma unroll
    for (int v = 0; v < 8; ++v) {
      const int k = kk * 32 + hi * 16 + 2 * v;
      a[kk].u[v] = *(const unsigned*)&Xs[(row0 + l15) * 128 + k];
    }
  }

  for (int nt = 0; nt < 8; ++nt) {
    const int n = nt * 16 + l15;     // output column (0..127), per lane
    const float bv = bias[n];
    v8f c;
#pragma unroll
    for (int r = 0; r < 8; ++r) c[r] = bv;

#pragma unroll
    for (int kk = 0; kk < 4; ++kk) {
      const v16h bf = *(const v16h*)&Ws[n * 128 + kk * 32 + hi * 16];
      c = wmma_f16(a[kk].h, bf, c);
    }

    // Branch-free store: resolve destination pointer + stride once.
    const int h  = n >> 5;           // head
    const int dh = n & 31;           // within-head dim
    const size_t bh   = (size_t)(b * H_ + h);
    const int    srow = s0 + row0 + 8 * hi;
    _Float16* dst; int stride;
    if (mode == 2) { dst = Vt + (bh * DH_ + dh) * S_ + srow; stride = 1;  }
    else {
      _Float16* base = (mode == 0) ? Qh : Kh;
      dst = base + (bh * S_ + srow) * DH_ + dh; stride = DH_;
    }
#pragma unroll
    for (int r = 0; r < 8; ++r) dst[r * stride] = (_Float16)c[r];
  }
}

// ---------------------------------------------------------------------------
// Kernel 2: flash attention. One block = 128 query rows of one (b,h);
// 8 waves x 16 rows. Online softmax over key tiles of 64.
// ---------------------------------------------------------------------------
__global__ __launch_bounds__(256) void flash_attn_kernel(
    const _Float16* __restrict__ Qh, const _Float16* __restrict__ Kh,
    const _Float16* __restrict__ Vt,
    const float* __restrict__ q_mask, const float* __restrict__ k_mask,
    const int* __restrict__ causal_p, float* __restrict__ out)
{
  __shared__ alignas(32) _Float16 Plds[8 * 16 * KT_];   // 2KB per wave

  const int qt = blockIdx.x, h = blockIdx.y, b = blockIdx.z;
  const int wave = threadIdx.x >> 5;
  const int lane = threadIdx.x & 31;
  const int l15  = lane & 15;
  const int hi   = lane >> 4;
  const int q0   = qt * 128 + wave * 16;
  const int causal = *causal_p;
  const int bS = b * S_;

  const _Float16* Qb = Qh + (size_t)(b * H_ + h) * S_ * DH_;
  const _Float16* Kb = Kh + (size_t)(b * H_ + h) * S_ * DH_;
  const _Float16* Vb = Vt + (size_t)(b * H_ + h) * DH_ * S_;

  // Q A-fragment: 16 rows x full dh=32 contraction, loaded once.
  AFrag qa;
#pragma unroll
  for (int v = 0; v < 8; ++v) {
    const int k = hi * 16 + 2 * v;
    qa.u[v] = *(const unsigned*)(Qb + (size_t)(q0 + l15) * DH_ + k);
  }

  float qm[8];
#pragma unroll
  for (int r = 0; r < 8; ++r) qm[r] = q_mask[bS + q0 + r + 8 * hi];

  float mrun[8], lrun[8];
#pragma unroll
  for (int r = 0; r < 8; ++r) { mrun[r] = -1e30f; lrun[r] = 0.0f; }
  v8f acc0 = {}; v8f acc1 = {};

  // Causal bound on 64-key tiles (block-uniform); always >= 1 tile so fully
  // masked rows still yield acc=0 / l>0 -> out=0 like the reference.
  long kend = (long)(qt * 128 + 127) + (long)causal + 1;
  int nkt = (int)(kend > 0 ? (kend + (KT_ - 1)) / KT_ : 1);
  if (nkt < 1) nkt = 1;
  if (nkt > S_ / KT_) nkt = S_ / KT_;

  _Float16* Pw = &Plds[wave * 16 * KT_];

  for (int kt = 0; kt < nkt; ++kt) {
    const int kc = kt * KT_;

    // 4 score WMMAs: 16 q-rows x 64 keys, full dh contraction each.
    v8f c[4];
    float km[4];
#pragma unroll
    for (int t = 0; t < 4; ++t) {
      const v16h kb =
          *(const v16h*)(Kb + (size_t)(kc + t * 16 + l15) * DH_ + hi * 16);
      v8f z = {};
      c[t] = wmma_f16(qa.h, kb, z);
      km[t] = k_mask[bS + kc + t * 16 + l15];
    }

#pragma unroll
    for (int r = 0; r < 8; ++r) {
      const int row = q0 + r + 8 * hi;
      float sv[4]; bool keep[4];
#pragma unroll
      for (int t = 0; t < 4; ++t) {
        const int col = kc + t * 16 + l15;
        keep[t] = (qm[r] * km[t] != 0.0f) && (col <= row + causal);
        // Reference: mask added BEFORE the 1/sqrt(dh) scaling.
        sv[t] = (c[t][r] + (keep[t] ? 0.0f : NEG_MASK)) * ISQ;
      }
      float mx = fmaxf(fmaxf(sv[0], sv[1]), fmaxf(sv[2], sv[3]));
      mx = fmaxf(mx, __shfl_xor(mx, 1, 32));
      mx = fmaxf(mx, __shfl_xor(mx, 2, 32));
      mx = fmaxf(mx, __shfl_xor(mx, 4, 32));
      mx = fmaxf(mx, __shfl_xor(mx, 8, 32));
      const float mnew  = fmaxf(mrun[r], mx);
      const float alpha = __expf(mrun[r] - mnew);
      float e[4];
      float rs = 0.0f;
#pragma unroll
      for (int t = 0; t < 4; ++t) { e[t] = __expf(sv[t] - mnew); rs += e[t]; }
      rs += __shfl_xor(rs, 1, 32);
      rs += __shfl_xor(rs, 2, 32);
      rs += __shfl_xor(rs, 4, 32);
      rs += __shfl_xor(rs, 8, 32);
      lrun[r] = lrun[r] * alpha + rs;       // denominator: UNmasked probs
      mrun[r] = mnew;
      acc0[r] *= alpha;
      acc1[r] *= alpha;
      // post-softmax mask multiply -> zero masked probs before PV
      const int rl = r + 8 * hi;
#pragma unroll
      for (int t = 0; t < 4; ++t)
        Pw[rl * KT_ + t * 16 + l15] = (_Float16)(keep[t] ? e[t] : 0.0f);
    }
    asm volatile("s_wait_dscnt 0" ::: "memory");   // P stores -> P frag reads

    // P A-fragments (16x64 -> two 16x32) read back in A layout.
    AFrag pa0, pa1;
#pragma unroll
    for (int v = 0; v < 8; ++v) {
      const int k = hi * 16 + 2 * v;
      pa0.u[v] = *(const unsigned*)&Pw[l15 * KT_ + k];
      pa1.u[v] = *(const unsigned*)&Pw[l15 * KT_ + 32 + k];
    }
    // V B-fragments from transposed V: contiguous 32B per lane.
    const v16h vb00 = *(const v16h*)(Vb + (size_t)(l15)      * S_ + kc +      hi * 16);
    const v16h vb01 = *(const v16h*)(Vb + (size_t)(16 + l15) * S_ + kc +      hi * 16);
    const v16h vb10 = *(const v16h*)(Vb + (size_t)(l15)      * S_ + kc + 32 + hi * 16);
    const v16h vb11 = *(const v16h*)(Vb + (size_t)(16 + l15) * S_ + kc + 32 + hi * 16);
    acc0 = wmma_f16(pa0.h, vb00, acc0);
    acc1 = wmma_f16(pa0.h, vb01, acc1);
    acc0 = wmma_f16(pa1.h, vb10, acc0);
    acc1 = wmma_f16(pa1.h, vb11, acc1);
  }

#pragma unroll
  for (int r = 0; r < 8; ++r) {
    const int s = q0 + r + 8 * hi;
    const float inv = 1.0f / lrun[r];
    out[(size_t)(bS + s) * D_ + h * DH_ + l15]      = acc0[r] * inv;
    out[(size_t)(bS + s) * D_ + h * DH_ + 16 + l15] = acc1[r] * inv;
  }
}

// ---------------------------------------------------------------------------
extern "C" void kernel_launch(void* const* d_in, const int* in_sizes, int n_in,
                              void* d_out, int out_size, void* d_ws, size_t ws_size,
                              hipStream_t stream) {
  const float* query = (const float*)d_in[0];
  const float* key   = (const float*)d_in[1];
  const float* value = (const float*)d_in[2];
  const float* q_mask = (const float*)d_in[3];
  const float* k_mask = (const float*)d_in[4];
  const float* WQ_w = (const float*)d_in[5];
  const float* WQ_b = (const float*)d_in[6];
  const float* WK_w = (const float*)d_in[7];
  const float* WK_b = (const float*)d_in[8];
  const float* WV_w = (const float*)d_in[9];
  const float* WV_b = (const float*)d_in[10];
  const int*   causal = (const int*)d_in[11];

  const size_t per = (size_t)B_ * H_ * S_ * DH_;   // 2M f16 elements each
  _Float16* Qh = (_Float16*)d_ws;
  _Float16* Kh = Qh + per;
  _Float16* Vt = Kh + per;

  dim3 gproj(B_ * S_ / 128, 3, 1);
  qkv_proj_kernel<<<gproj, 256, 0, stream>>>(query, key, value,
                                             WQ_w, WQ_b, WK_w, WK_b, WV_w, WV_b,
                                             Qh, Kh, Vt);

  dim3 gattn(S_ / 128, H_, B_);
  flash_attn_kernel<<<gattn, 256, 0, stream>>>(Qh, Kh, Vt, q_mask, k_mask,
                                               causal, (float*)d_out);
}
